// DiceLoss_76854144795002
// MI455X (gfx1250) — compile-verified, matrix operand order checked by reading, not run on hardware
//
#include <hip/hip_runtime.h>
#include <hip/hip_bf16.h>

typedef __attribute__((ext_vector_type(4)))  float    f32x4;
typedef __attribute__((ext_vector_type(4)))  int      i32x4;
typedef __attribute__((ext_vector_type(16))) _Float16 v16h;
typedef __attribute__((ext_vector_type(8)))  float    v8f;

// Problem geometry (fixed by the reference).
constexpr int       kC       = 5;                       // classes
constexpr long long kS       = 96LL * 192 * 192;        // spatial per batch = 3,538,944
constexpr long long kN       = 2 * kS;                  // total elements    = 7,077,888
constexpr int       kThreads = 256;
constexpr int       kEPT     = 4;                       // elements per thread
constexpr int       kNB      = (int)(kN / ((long long)kThreads * kEPT)); // 6912 blocks
static_assert(kN == (long long)kNB * kThreads * kEPT, "exact tiling");
static_assert(kNB % 32 == 0, "WMMA K-tiling needs multiple of 32");

// ---------------------------------------------------------------------------
// Kernel 1: streaming pass. Each thread handles 4 consecutive spatial elements,
// loads 5 logit planes as b128 (non-temporal, single-use data), computes
// argmax + per-class indicator counts, reduces over the wave32 with shuffles,
// then over the block with 8 LDS atomics per wave. Block partials (<=1024, so
// exact in f16) are written as a 16 x kNB f16 matrix: rows 0..3 = tsum(c=1..4),
// rows 4..7 = inter(c=1..4), rows 8..15 = zero padding for the WMMA A-matrix.
// ---------------------------------------------------------------------------
__global__ __launch_bounds__(kThreads)
void dice_partial_kernel(const float* __restrict__ logits,
                         const int*   __restrict__ tgt,
                         _Float16*    __restrict__ part) {
  __shared__ unsigned sh[8];
  const int tid = threadIdx.x;
  if (tid < 8) sh[tid] = 0u;
  __syncthreads();

  const long long e0 = ((long long)blockIdx.x * kThreads + tid) * kEPT;
  const int       b  = (int)(e0 / kS);         // all 4 elements share one batch (kS % 4 == 0)
  const long long s  = e0 - (long long)b * kS;
  const long long pb = (long long)b * kC * kS + s;

  f32x4 v[kC];
#pragma unroll
  for (int c = 0; c < kC; ++c) {
    const f32x4* p = (const f32x4*)(logits + pb + (long long)c * kS); // 16B aligned
    v[c] = __builtin_nontemporal_load(p);
  }
  const i32x4 t4 = __builtin_nontemporal_load((const i32x4*)(tgt + e0));

  int cnt[8];
#pragma unroll
  for (int k = 0; k < 8; ++k) cnt[k] = 0;

#pragma unroll
  for (int j = 0; j < kEPT; ++j) {
    float best = v[0][j];
    int   m    = 0;
#pragma unroll
    for (int c = 1; c < kC; ++c) {          // strict '>' == first-max (jnp.argmax)
      const float lv = v[c][j];
      if (lv > best) { best = lv; m = c; }
    }
    const int t = t4[j];
#pragma unroll
    for (int c = 1; c < kC; ++c) {          // static indices only -> stays in VGPRs
      const int te = (t == c) ? 1 : 0;
      cnt[c - 1] += te;                     // tsum[c]
      cnt[c + 3] += te & ((m == c) ? 1 : 0);// inter[c]
    }
  }

  // wave32 butterfly reduction
#pragma unroll
  for (int k = 0; k < 8; ++k) {
    int x = cnt[k];
#pragma unroll
    for (int off = 16; off > 0; off >>= 1) x += __shfl_xor(x, off, 32);
    cnt[k] = x;
  }
  if ((tid & 31) == 0) {
#pragma unroll
    for (int k = 0; k < 8; ++k) atomicAdd(&sh[k], (unsigned)cnt[k]);
  }
  __syncthreads();

  if (tid == 0) {
#pragma unroll
    for (int k = 0; k < 8; ++k)
      part[(long long)k * kNB + blockIdx.x] = (_Float16)(float)sh[k]; // <=1024: exact f16
#pragma unroll
    for (int k = 8; k < 16; ++k)
      part[(long long)k * kNB + blockIdx.x] = (_Float16)0.0f;          // zero rows for WMMA
  }
}

// ---------------------------------------------------------------------------
// Kernel 2: one wave. Reduce the 16 x kNB partial matrix with
// v_wmma_f32_16x16x32_f16 against an all-ones B: D[m][n] = sum_k A[m][k].
// f32 accumulation of integer-valued products is exact (totals < 2^24).
// A layout (16-bit, 16x32): lane L -> row L%16; K-base = (L<16) ? 0 : 8;
// a[0..7] = K base+0..7, a[8..15] = K base+16..23.
// ---------------------------------------------------------------------------
__global__ __launch_bounds__(32)
void dice_final_kernel(const _Float16* __restrict__ part,
                       float*          __restrict__ out) {
  const int lane  = threadIdx.x;
  const int m     = lane & 15;
  const int kbase = (lane < 16) ? 0 : 8;
  const _Float16* row = part + (long long)m * kNB + kbase;

  v16h bones;
#pragma unroll
  for (int i = 0; i < 16; ++i) bones[i] = (_Float16)1.0f;

  v8f acc = {};
  for (int j0 = 0; j0 < kNB; j0 += 32) {
    const _Float16* rp = row + j0;          // 16B aligned
    v16h a;
#pragma unroll
    for (int vv = 0; vv < 8; ++vv) a[vv] = rp[vv];          // K = base+0..7
#pragma unroll
    for (int vv = 0; vv < 8; ++vv) a[8 + vv] = rp[16 + vv]; // K = base+16..23
    acc = __builtin_amdgcn_wmma_f32_16x16x32_f16(
        /*neg_a=*/false, a, /*neg_b=*/false, bones,
        /*c_mod=*/(short)0, acc, /*reuse_a=*/false, /*reuse_b=*/false);
  }

  if (lane == 0) {
    // lane 0, VGPR r -> D[M=r][N=0]; rows 0..3 = tsum(1..4), rows 4..7 = inter(1..4)
    const float EPS = 1e-8f;
    float dsum = 0.0f;
#pragma unroll
    for (int c = 0; c < 4; ++c) {
      const float ts = acc[c];
      const float in = acc[c + 4];
      dsum += (2.0f * in + EPS) / (in + ts + EPS);
    }
    out[0] = 1.0f - 0.25f * dsum;
  }
}

extern "C" void kernel_launch(void* const* d_in, const int* in_sizes, int n_in,
                              void* d_out, int out_size, void* d_ws, size_t ws_size,
                              hipStream_t stream) {
  (void)in_sizes; (void)n_in; (void)out_size; (void)ws_size;
  const float* logits = (const float*)d_in[0];
  const int*   tgt    = (const int*)d_in[1];
  _Float16*    part   = (_Float16*)d_ws;     // needs 16*6912*2 = 221,184 bytes
  float*       out    = (float*)d_out;

  dice_partial_kernel<<<kNB, kThreads, 0, stream>>>(logits, tgt, part);
  dice_final_kernel<<<1, 32, 0, stream>>>(part, out);
}
